// PersonReIDLoss_61572651155654
// MI455X (gfx1250) — matrix operand order repaired and checked by reading, not compiled, not run on hardware
//
#include <hip/hip_runtime.h>

// ---------------------------------------------------------------------------
// PersonReID triplet loss, fused for MI455X (gfx1250, wave32, WMMA).
// features: [B=4096, D=1024] f32, labels: [B] i32  ->  scalar f32 loss.
//
// Two 4096x4096x1024 gram GEMMs (P@F^T, N@F^T) on v_wmma_f32_16x16x32_bf16,
// f32 accumulation, fused sqrt/relu/mean epilogue (no [B,B] intermediates).
// Register-blocked: each wave owns 32x32 output for BOTH gemms ->
// 8 WMMAs per 6 operand loads per k-step (L2-traffic bound otherwise).
// ---------------------------------------------------------------------------

typedef __attribute__((ext_vector_type(16))) __bf16   v16bf;
typedef __attribute__((ext_vector_type(8)))  float    v8f;
typedef __attribute__((ext_vector_type(4)))  unsigned u32x4;

#define RB 4096           // batch
#define RD 1024           // feature dim
#define MARGIN_F 0.3f
#define EPS_F 1e-6f

// ---------------- helpers ----------------

__device__ __forceinline__ unsigned short f32_to_bf16_rne(float f) {
    union { float f; unsigned u; } x; x.f = f;
    unsigned u = x.u;
    unsigned r = u + 0x7FFFu + ((u >> 16) & 1u);     // round to nearest even
    if ((u & 0x7F800000u) == 0x7F800000u) r = u;     // inf/nan passthrough
    return (unsigned short)(r >> 16);
}

__device__ __forceinline__ unsigned mix_hash(unsigned a, unsigned b, unsigned salt) {
    unsigned h = a * 0x9E3779B1u ^ b * 0x85EBCA77u ^ salt;
    h ^= h >> 16; h *= 0x7FEB352Du;
    h ^= h >> 15; h *= 0x846CA68Bu;
    h ^= h >> 16;
    return h;
}

// load 16 bf16 (two aligned 16B chunks) into a WMMA operand register set
__device__ __forceinline__ v16bf load_op16(const unsigned short* p0,
                                           const unsigned short* p1) {
    union { v16bf v; u32x4 q[2]; } u;
    u.q[0] = *reinterpret_cast<const u32x4*>(p0);   // global_load_b128
    u.q[1] = *reinterpret_cast<const u32x4*>(p1);   // global_load_b128
    return u.v;
}

__device__ __forceinline__ v8f wmma_bf16(v16bf a, v16bf b, v8f c) {
    return __builtin_amdgcn_wmma_f32_16x16x32_bf16(false, a, false, b,
                                                   (short)0, c, false, false);
}

// ---------------- K1: per-row stats + f32 -> bf16 convert ----------------

__global__ void __launch_bounds__(256)
row_stats_convert_kernel(const float* __restrict__ F,
                         unsigned short* __restrict__ Abf,
                         float* __restrict__ sumF,
                         float* __restrict__ nrmF) {
    __shared__ float s_sum[256];
    __shared__ float s_sq[256];
    const int row = blockIdx.x;
    const int tid = threadIdx.x;
    const size_t base = (size_t)row * RD;
    float s = 0.f, q = 0.f;
#pragma unroll
    for (int c = 0; c < RD; c += 256) {
        float f = F[base + c + tid];
        Abf[base + c + tid] = f32_to_bf16_rne(f);
        s += f;
        q += f * f;
    }
    s_sum[tid] = s; s_sq[tid] = q;
    __syncthreads();
    for (int off = 128; off > 0; off >>= 1) {
        if (tid < off) { s_sum[tid] += s_sum[tid + off]; s_sq[tid] += s_sq[tid + off]; }
        __syncthreads();
    }
    if (tid == 0) { sumF[row] = s_sum[0]; nrmF[row] = s_sq[0]; }
}

// ---------------- K2: triplet mining (deterministic hash-argmax) ----------------

__global__ void __launch_bounds__(256)
sample_triplets_kernel(const int* __restrict__ labels,
                       int* __restrict__ posIdx,
                       int* __restrict__ negIdx) {
    __shared__ int s_lab[RB];
    const int tid = threadIdx.x;
    for (int j = tid; j < RB; j += 256) s_lab[j] = labels[j];
    __syncthreads();

    const int i = blockIdx.x * 256 + tid;
    const int li = s_lab[i];
    unsigned bp = 0u, bn = 0u;
    int pos = i, neg = -1;
    for (int j = 0; j < RB; ++j) {
        if (s_lab[j] == li) {
            unsigned h = mix_hash((unsigned)i, (unsigned)j, 0x68BC21EBu);
            if (h >= bp) { bp = h; pos = j; }
        } else {
            unsigned h = mix_hash((unsigned)i, (unsigned)j, 0x02E5BE93u);
            if (h >= bn) { bn = h; neg = j; }
        }
    }
    if (neg < 0) neg = (int)(mix_hash((unsigned)i, 0u, 0xDEADBEEFu) & (RB - 1));
    posIdx[i] = pos;
    negIdx[i] = neg;
}

// ---------------- K3: fused dual-GEMM (WMMA bf16) + distance/relu epilogue ----
// Block = 256 threads = 8 waves; block tile = 128 rows(i) x 64 cols(j).
// Wave w: wi = w>>1 (0..3, 32 rows each), wj = w&1 (0..1, 32 cols each).
// Each wave: 2x2 grid of 16x16 WMMA tiles for P-gemm AND N-gemm
// -> 8 v_wmma per k-step against 6 operand loads (a_p0,a_p1,a_n0,a_n1,b0,b1).

__global__ void __launch_bounds__(256)
triplet_wmma_loss_kernel(const unsigned short* __restrict__ Abf,
                         const int* __restrict__ posIdx,
                         const int* __restrict__ negIdx,
                         const float* __restrict__ sumF,
                         const float* __restrict__ nrmF,
                         float* __restrict__ partials) {
    __shared__ int   s_pi[128], s_ni[128];
    __shared__ float s_nrmP[128], s_sumP[128], s_nrmN[128], s_sumN[128];
    __shared__ float s_red[256];

    const int tid  = threadIdx.x;
    const int wave = tid >> 5;
    const int lane = tid & 31;
    const int wi   = wave >> 1;     // 0..3  -> 32-row i sub-tile
    const int wj   = wave & 1;      // 0..1  -> 32-col j sub-tile
    const int i0   = blockIdx.y * 128;
    const int j0   = blockIdx.x * 64;

    if (tid < 128) {
        int pi = posIdx[i0 + tid];
        int ni = negIdx[i0 + tid];
        s_pi[tid] = pi;            s_ni[tid] = ni;
        s_nrmP[tid] = nrmF[pi];    s_sumP[tid] = sumF[pi];
        s_nrmN[tid] = nrmF[ni];    s_sumN[tid] = sumF[ni];
    }
    __syncthreads();

    const int mlane = lane & 15;
    const int r0 = wi * 32 + mlane;         // local i-row, M-subtile 0
    const int r1 = r0 + 16;                 // local i-row, M-subtile 1
    const int jg0 = j0 + wj * 32 + mlane;   // global j-col, N-subtile 0
    const int jg1 = jg0 + 16;               // global j-col, N-subtile 1

    const unsigned short* Ap0 = Abf + (size_t)s_pi[r0] * RD;
    const unsigned short* Ap1 = Abf + (size_t)s_pi[r1] * RD;
    const unsigned short* An0 = Abf + (size_t)s_ni[r0] * RD;
    const unsigned short* An1 = Abf + (size_t)s_ni[r1] * RD;
    const unsigned short* Bf0 = Abf + (size_t)jg0 * RD;
    const unsigned short* Bf1 = Abf + (size_t)jg1 * RD;

    // ISA 7.12.2 lane->K maps (16-bit operands, wave32):
    //   A 16x32: lanes<16 hold K {0..7,16..23}, lanes>=16 hold K {8..15,24..31}
    //   B 32x16: lanes<16 hold K {0..7,16..23}, lanes>=16 hold K {8..15,24..31}
    const int offA0 = (lane < 16) ? 0 : 8;  const int offA1 = offA0 + 16;
    const int offB0 = (lane < 16) ? 0 : 8;  const int offB1 = offB0 + 16;

    v8f cp00 = {}, cp01 = {}, cp10 = {}, cp11 = {};   // P @ F^T, [Msub][Nsub]
    v8f cn00 = {}, cn01 = {}, cn10 = {}, cn11 = {};   // N @ F^T

#pragma unroll 2
    for (int k0 = 0; k0 < RD; k0 += 32) {
        v16bf ap0 = load_op16(Ap0 + k0 + offA0, Ap0 + k0 + offA1);
        v16bf ap1 = load_op16(Ap1 + k0 + offA0, Ap1 + k0 + offA1);
        v16bf an0 = load_op16(An0 + k0 + offA0, An0 + k0 + offA1);
        v16bf an1 = load_op16(An1 + k0 + offA0, An1 + k0 + offA1);
        v16bf b0  = load_op16(Bf0 + k0 + offB0, Bf0 + k0 + offB1);
        v16bf b1  = load_op16(Bf1 + k0 + offB0, Bf1 + k0 + offB1);

        cp00 = wmma_bf16(ap0, b0, cp00);
        cp01 = wmma_bf16(ap0, b1, cp01);
        cp10 = wmma_bf16(ap1, b0, cp10);
        cp11 = wmma_bf16(ap1, b1, cp11);
        cn00 = wmma_bf16(an0, b0, cn00);
        cn01 = wmma_bf16(an0, b1, cn01);
        cn10 = wmma_bf16(an1, b0, cn10);
        cn11 = wmma_bf16(an1, b1, cn11);
    }

    // epilogue: gram trick -> distances -> hinge, per C-layout element
    const float nj0 = nrmF[jg0], sj0 = sumF[jg0];
    const float nj1 = nrmF[jg1], sj1 = sumF[jg1];
    const float DEPS2 = (float)RD * EPS_F * EPS_F;
    float acc = 0.f;
#pragma unroll
    for (int v = 0; v < 8; ++v) {
        const int M  = v + ((lane < 16) ? 0 : 8);
        const int ra = wi * 32 + M;          // local stats row, M-subtile 0
        const int rb = ra + 16;              // local stats row, M-subtile 1
        const float nP0 = s_nrmP[ra], sP0 = s_sumP[ra];
        const float nP1 = s_nrmP[rb], sP1 = s_sumP[rb];
        const float nN0 = s_nrmN[ra], sN0 = s_sumN[ra];
        const float nN1 = s_nrmN[rb], sN1 = s_sumN[rb];

        float d;
        // (Msub0, Nsub0)
        d  = fmaxf(sqrtf(fmaxf(nj0 + nP0 - 2.f*cp00[v] + 2.f*EPS_F*(sj0 - sP0) + DEPS2, 1e-12f))
                 - sqrtf(fmaxf(nj0 + nN0 - 2.f*cn00[v] + 2.f*EPS_F*(sj0 - sN0) + DEPS2, 1e-12f))
                 + MARGIN_F, 0.f);
        acc += d;
        // (Msub0, Nsub1)
        d  = fmaxf(sqrtf(fmaxf(nj1 + nP0 - 2.f*cp01[v] + 2.f*EPS_F*(sj1 - sP0) + DEPS2, 1e-12f))
                 - sqrtf(fmaxf(nj1 + nN0 - 2.f*cn01[v] + 2.f*EPS_F*(sj1 - sN0) + DEPS2, 1e-12f))
                 + MARGIN_F, 0.f);
        acc += d;
        // (Msub1, Nsub0)
        d  = fmaxf(sqrtf(fmaxf(nj0 + nP1 - 2.f*cp10[v] + 2.f*EPS_F*(sj0 - sP1) + DEPS2, 1e-12f))
                 - sqrtf(fmaxf(nj0 + nN1 - 2.f*cn10[v] + 2.f*EPS_F*(sj0 - sN1) + DEPS2, 1e-12f))
                 + MARGIN_F, 0.f);
        acc += d;
        // (Msub1, Nsub1)
        d  = fmaxf(sqrtf(fmaxf(nj1 + nP1 - 2.f*cp11[v] + 2.f*EPS_F*(sj1 - sP1) + DEPS2, 1e-12f))
                 - sqrtf(fmaxf(nj1 + nN1 - 2.f*cn11[v] + 2.f*EPS_F*(sj1 - sN1) + DEPS2, 1e-12f))
                 + MARGIN_F, 0.f);
        acc += d;
    }

    s_red[tid] = acc;
    __syncthreads();
    for (int off = 128; off > 0; off >>= 1) {
        if (tid < off) s_red[tid] += s_red[tid + off];
        __syncthreads();
    }
    if (tid == 0) partials[blockIdx.y * gridDim.x + blockIdx.x] = s_red[0];
}

// ---------------- K4: deterministic final reduction -> mean ----------------

__global__ void __launch_bounds__(256)
finalize_kernel(const float* __restrict__ partials, int n, float* __restrict__ out) {
    __shared__ float s_red[256];
    const int tid = threadIdx.x;
    float acc = 0.f;
    for (int k = tid; k < n; k += 256) acc += partials[k];
    s_red[tid] = acc;
    __syncthreads();
    for (int off = 128; off > 0; off >>= 1) {
        if (tid < off) s_red[tid] += s_red[tid + off];
        __syncthreads();
    }
    if (tid == 0) out[0] = s_red[0] * (1.f / ((float)RB * (float)RB));
}

// ---------------- launch ----------------

extern "C" void kernel_launch(void* const* d_in, const int* in_sizes, int n_in,
                              void* d_out, int out_size, void* d_ws, size_t ws_size,
                              hipStream_t stream) {
    const float* features = (const float*)d_in[0];   // [4096,1024] f32
    const int*   labels   = (const int*)d_in[1];     // [4096] i32
    float* out = (float*)d_out;

    // workspace layout (all 256B-aligned)
    char* ws = (char*)d_ws;
    int*   posIdx = (int*)(ws);                                  // 16 KB
    int*   negIdx = (int*)(ws + 16384);                          // 16 KB
    float* sumF   = (float*)(ws + 32768);                        // 16 KB
    float* nrmF   = (float*)(ws + 49152);                        // 16 KB
    unsigned short* Abf = (unsigned short*)(ws + 65536);         // 8 MB bf16
    float* partials = (float*)(ws + 65536 + (size_t)RB * RD * 2);// <= 32 KB

    row_stats_convert_kernel<<<RB, 256, 0, stream>>>(features, Abf, sumF, nrmF);
    sample_triplets_kernel<<<RB / 256, 256, 0, stream>>>(labels, posIdx, negIdx);

    dim3 grid(RB / 64, RB / 128);   // (64, 32) -> 2048 tiles of 128x64
    triplet_wmma_loss_kernel<<<grid, 256, 0, stream>>>(Abf, posIdx, negIdx,
                                                       sumF, nrmF, partials);

    finalize_kernel<<<1, 256, 0, stream>>>(partials, (int)(grid.x * grid.y), out);
}
